// GeoLocalizationNet_79663053406285
// MI455X (gfx1250) — compile-verified, matrix operand order checked by reading, not run on hardware
//
#include <hip/hip_runtime.h>

typedef __attribute__((ext_vector_type(16))) __bf16 v16bf;
typedef __attribute__((ext_vector_type(8)))  __bf16 v8bf;
typedef __attribute__((ext_vector_type(8)))  float  v8f;

#define N_  16
#define D_  2048
#define HW_ 1024
#define F_  512

// ---------------------------------------------------------------------------
// CDNA5 async copy helpers (ASYNCcnt-tracked DMA global->LDS, ISA §15.18.3)
// ---------------------------------------------------------------------------
__device__ __forceinline__ void async_b128(void* lds, const void* g) {
  unsigned l = (unsigned)(unsigned long long)(uintptr_t)lds;  // LDS byte offset
  unsigned long long ga = (unsigned long long)(uintptr_t)g;
  asm volatile("global_load_async_to_lds_b128 %0, %1, off"
               :: "v"(l), "v"(ga) : "memory");
}
__device__ __forceinline__ void wait_async0() {
  asm volatile("s_wait_asynccnt 0x0" ::: "memory");
}

// ---------------------------------------------------------------------------
// k_prep: convert proj_w -> bf16, zero gem accumulator and norm accumulator
// ---------------------------------------------------------------------------
__global__ void k_prep(const float* __restrict__ w, __bf16* __restrict__ wbf,
                       float* __restrict__ gem, float* __restrict__ nrm) {
  int idx = blockIdx.x * 256 + threadIdx.x;          // 0 .. 1048575 == F_*D_
  wbf[idx] = (__bf16)w[idx];
  if (idx < N_ * D_)  gem[idx] = 0.f;
  if (idx < N_ * HW_) nrm[idx] = 0.f;
}

// ---------------------------------------------------------------------------
// k_norm1: partial channel sum-of-squares, split 8 ways over d for parallelism
// ---------------------------------------------------------------------------
__global__ void k_norm1(const float* __restrict__ x, float* __restrict__ nrm) {
  int b  = blockIdx.x;                 // 512 blocks
  int pb = b >> 3, ds = b & 7;
  int pos = pb * 256 + threadIdx.x;    // 0..16383  (n*1024 + hw)
  int n = pos >> 10, p = pos & 1023;
  const float* xb = x + ((size_t)n * D_) * HW_ + p;
  int d0 = ds * 256;
  float a = 0.f;
  for (int d = 0; d < 256; ++d) {
    float v = xb[(size_t)(d0 + d) * HW_];
    a = fmaf(v, v, a);
  }
  atomicAdd(&nrm[pos], a);
}

// k_norm2: sumsq -> 1/max(sqrt(s),1e-12) in place
__global__ void k_norm2(float* __restrict__ nrm) {
  int i = blockIdx.x * 256 + threadIdx.x;
  nrm[i] = 1.f / fmaxf(sqrtf(nrm[i]), 1e-12f);
}

// ---------------------------------------------------------------------------
// k_box: normalized x -> (a) GeM clip^p accumulation, (b) 3x3 box-sum written
// transposed as bf16 A_t[(n*HW+pos)][d].  Block = (n, 32-d chunk, 16x16 tile
// with 1-px halo).  Out-of-image halo entries are ZERO (matches the reference
// integral-image boundary semantics exactly).
// ---------------------------------------------------------------------------
#define DB 32
#define XPITCH 326   // 18*18=324 rounded to odd-word pitch -> conflict-free
__global__ void k_box(const float* __restrict__ x, const float* __restrict__ nrm,
                      const float* __restrict__ pp, __bf16* __restrict__ At,
                      float* __restrict__ gem) {
  __shared__ __bf16 xn[DB * XPITCH];
  __shared__ float  gp[DB];
  int b = blockIdx.x;                       // 16*64*4 = 4096 blocks
  int tile = b & 3, dc = (b >> 2) & 63, n = b >> 8;
  int h0 = (tile >> 1) * 16, w0 = (tile & 1) * 16;
  int d0 = dc * DB;
  int t  = threadIdx.x;
  float pv = pp[0];
  if (t < DB) gp[t] = 0.f;
  __syncthreads();
  const float* xb = x + ((size_t)n * D_ + d0) * HW_;
  const float* nb = nrm + n * HW_;
  for (int d = 0; d < DB; ++d) {
    float lsum = 0.f;
    for (int i = t; i < 324; i += 256) {     // 18x18 halo tile
      int r = i / 18, c = i - r * 18;
      int hh = h0 - 1 + r, ww = w0 - 1 + c;
      float v = 0.f;
      if (hh >= 0 && hh < 32 && ww >= 0 && ww < 32)
        v = xb[(size_t)d * HW_ + hh * 32 + ww] * nb[hh * 32 + ww];
      xn[d * XPITCH + i] = (__bf16)v;
      // GeM contribution only from owned (interior) positions, in f32
      if (r >= 1 && r <= 16 && c >= 1 && c <= 16) {
        float cc = fmaxf(v, 1e-6f);
        lsum += __powf(cc, pv);
      }
    }
    for (int m = 16; m >= 1; m >>= 1) lsum += __shfl_xor(lsum, m, 32);
    if ((t & 31) == 0) atomicAdd(&gp[d], lsum);
  }
  __syncthreads();
  if (t < DB) atomicAdd(&gem[n * D_ + d0 + t], gp[t]);

  // box phase: lanes span d (coalesced 64B transposed stores per position)
  int j = t & 31, rg = t >> 5;
  const __bf16* xj = xn + j * XPITCH;
  for (int it = 0; it < 32; ++it) {
    int pidx = it * 8 + rg;                 // 0..255 owned positions
    int pr = pidx >> 4, pc = pidx & 15;
    float s = 0.f;
#pragma unroll
    for (int dr = 0; dr < 3; ++dr)
#pragma unroll
      for (int dcc = 0; dcc < 3; ++dcc)
        s += (float)xj[(pr + dr) * 18 + pc + dcc];
    int pos = (h0 + pr) * 32 + (w0 + pc);
    At[((size_t)(n * HW_ + pos)) * D_ + d0 + j] = (__bf16)s;
  }
}

// ---------------------------------------------------------------------------
// k_gem: GeM finalize + [16,2048]x[2048,512] projection + bias + L2 norm
// ---------------------------------------------------------------------------
__global__ void k_gem(const float* __restrict__ gem, const float* __restrict__ w,
                      const float* __restrict__ bvec, const float* __restrict__ pp,
                      float* __restrict__ out) {
  __shared__ float row[D_];
  __shared__ float wsum[8];
  int n = blockIdx.x, t = threadIdx.x;
  float pv = pp[0];
  for (int i = t; i < D_; i += 256) {
    float s = gem[n * D_ + i] * (1.0f / 1024.0f);
    row[i] = __powf(s, 1.0f / pv);
  }
  __syncthreads();
  float o0 = bvec[t], o1 = bvec[t + 256];
  const float* w0 = w + (size_t)t * D_;
  const float* w1 = w + (size_t)(t + 256) * D_;
  for (int d = 0; d < D_; ++d) {
    float r = row[d];
    o0 = fmaf(r, w0[d], o0);
    o1 = fmaf(r, w1[d], o1);
  }
  float ss = o0 * o0 + o1 * o1;
  for (int m = 16; m >= 1; m >>= 1) ss += __shfl_xor(ss, m, 32);
  if ((t & 31) == 0) wsum[t >> 5] = ss;
  __syncthreads();
  float tot = 0.f;
#pragma unroll
  for (int i = 0; i < 8; ++i) tot += wsum[i];
  float inv = 1.f / fmaxf(sqrtf(tot), 1e-12f);
  out[n * F_ + t]       = o0 * inv;
  out[n * F_ + t + 256] = o1 * inv;
}

// ---------------------------------------------------------------------------
// k_proj: WMMA GEMM  C[16384,512] = A_t[16384,2048](bf16) * Wbf^T  + bias,
// then per-row L2 normalize, store to [N,F,HW] layout.
// Block: 64 M-rows x all 512 N.  8 waves = 2(M) x 4(N); wave tile 32x128.
// Double-buffered LDS fed by ASYNCcnt-tracked global_load_async_to_lds_b128.
// ---------------------------------------------------------------------------
#define APITCH 40            // 32 bf16 + pad; 80B rows keep 16B LDS alignment
#define STAGE_ELEMS ((64 + F_) * APITCH)    // A tile + B tile per stage
__global__ __launch_bounds__(256) void k_proj(const __bf16* __restrict__ At,
                                              const __bf16* __restrict__ Wbf,
                                              const float* __restrict__ bvec,
                                              float* __restrict__ outl) {
  extern __shared__ __align__(16) __bf16 smem[];
  __shared__ float rss[64];
  int t = threadIdx.x;
  int wave = t >> 5, lane = t & 31;
  int lm = lane & 15, lh = lane >> 4;
  int kb2 = lh ? 8 : 0;                 // ISA bf16 A/B per-lane K split
  int waveM = wave >> 2, waveN = wave & 3;
  int m0w = waveM * 32, n0w = waveN * 128;
  int rowBase = blockIdx.x * 64;        // global row = n*1024 + pos
  if (t < 64) rss[t] = 0.f;

  // per-thread copy slots (A: 1x b128 chunk, B: 8x b128 chunks per stage)
  int arow = t >> 2, ach = t & 3;
  unsigned aSlot = (unsigned)(arow * APITCH + ach * 8);       // elem offset
  const __bf16* aSrc = At + ((size_t)(rowBase + arow)) * D_ + ach * 8;

  v8f c[2][8];
#pragma unroll
  for (int a = 0; a < 2; ++a)
#pragma unroll
    for (int bq = 0; bq < 8; ++bq)
#pragma unroll
      for (int e = 0; e < 8; ++e) c[a][bq][e] = 0.f;

  // issue stage 0 (stage offset 0)
  {
    async_b128(smem + aSlot, aSrc);
#pragma unroll
    for (int i = 0; i < 8; ++i) {
      int idx = t + i * 256;
      int row = idx >> 2, ch = idx & 3;
      async_b128(smem + (64 * APITCH) + row * APITCH + ch * 8,
                 Wbf + (size_t)row * D_ + ch * 8);
    }
  }

  for (int kb = 0; kb < 64; ++kb) {
    unsigned curOff  = (kb & 1) ? (unsigned)STAGE_ELEMS : 0u;
    unsigned nextOff = (kb & 1) ? 0u : (unsigned)STAGE_ELEMS;
    wait_async0();        // my stage-kb DMAs have landed in LDS
    __syncthreads();      // everyone's landed; prev consumers of next buf done
    if (kb + 1 < 64) {
      int k0n = (kb + 1) * 32;
      async_b128(smem + nextOff + aSlot, aSrc + k0n);
#pragma unroll
      for (int i = 0; i < 8; ++i) {
        int idx = t + i * 256;
        int row = idx >> 2, ch = idx & 3;
        async_b128(smem + nextOff + (64 * APITCH) + row * APITCH + ch * 8,
                   Wbf + (size_t)row * D_ + k0n + ch * 8);
      }
    }
    const __bf16* As = smem + curOff;
    const __bf16* Bs = smem + curOff + 64 * APITCH;

    v16bf bf[8];
#pragma unroll
    for (int tn = 0; tn < 8; ++tn) {
      int col = n0w + tn * 16 + lm;
      v8bf lo = *(const v8bf*)&Bs[col * APITCH + kb2];
      v8bf hi = *(const v8bf*)&Bs[col * APITCH + kb2 + 16];
#pragma unroll
      for (int e = 0; e < 8; ++e) { bf[tn][e] = lo[e]; bf[tn][e + 8] = hi[e]; }
    }
#pragma unroll
    for (int tm = 0; tm < 2; ++tm) {
      int rowA = m0w + tm * 16 + lm;
      v8bf lo = *(const v8bf*)&As[rowA * APITCH + kb2];
      v8bf hi = *(const v8bf*)&As[rowA * APITCH + kb2 + 16];
      v16bf af;
#pragma unroll
      for (int e = 0; e < 8; ++e) { af[e] = lo[e]; af[e + 8] = hi[e]; }
#pragma unroll
      for (int tn = 0; tn < 8; ++tn) {
        c[tm][tn] = __builtin_amdgcn_wmma_f32_16x16x32_bf16(
            false, af, false, bf[tn], (short)0, c[tm][tn], false, false);
      }
    }
    __syncthreads();      // stage-kb consumption done before buf reused
  }

  // bias add + per-row sum of squares (rows = M, span F across waveN)
  float bv[8];
#pragma unroll
  for (int tn = 0; tn < 8; ++tn) bv[tn] = bvec[n0w + tn * 16 + lm];
#pragma unroll
  for (int tm = 0; tm < 2; ++tm) {
#pragma unroll
    for (int v = 0; v < 8; ++v) {
      float pvs = 0.f;
#pragma unroll
      for (int tn = 0; tn < 8; ++tn) {
        float val = c[tm][tn][v] + bv[tn];
        c[tm][tn][v] = val;
        pvs = fmaf(val, val, pvs);
      }
      pvs += __shfl_xor(pvs, 1, 32);
      pvs += __shfl_xor(pvs, 2, 32);
      pvs += __shfl_xor(pvs, 4, 32);
      pvs += __shfl_xor(pvs, 8, 32);
      if (lm == 0) atomicAdd(&rss[m0w + tm * 16 + lh * 8 + v], pvs);
    }
  }
  __syncthreads();

  int n = rowBase >> 10;
  int posB = rowBase & 1023;
  float* outn = outl + (size_t)n * F_ * HW_;
#pragma unroll
  for (int tm = 0; tm < 2; ++tm) {
    int rloc = m0w + tm * 16 + lh * 8;
    float inv[8];
#pragma unroll
    for (int v = 0; v < 8; ++v)
      inv[v] = 1.f / fmaxf(sqrtf(rss[rloc + v]), 1e-12f);
#pragma unroll
    for (int tn = 0; tn < 8; ++tn) {
      int f = n0w + tn * 16 + lm;
      float* dst = outn + (size_t)f * HW_ + posB + rloc;
      float4 s0 = make_float4(c[tm][tn][0] * inv[0], c[tm][tn][1] * inv[1],
                              c[tm][tn][2] * inv[2], c[tm][tn][3] * inv[3]);
      float4 s1 = make_float4(c[tm][tn][4] * inv[4], c[tm][tn][5] * inv[5],
                              c[tm][tn][6] * inv[6], c[tm][tn][7] * inv[7]);
      *(float4*)dst = s0;
      *(float4*)(dst + 4) = s1;
    }
  }
}

// ---------------------------------------------------------------------------
extern "C" void kernel_launch(void* const* d_in, const int* in_sizes, int n_in,
                              void* d_out, int out_size, void* d_ws, size_t ws_size,
                              hipStream_t stream) {
  const float* x  = (const float*)d_in[0];
  const float* w  = (const float*)d_in[1];
  const float* bb = (const float*)d_in[2];
  const float* p  = (const float*)d_in[3];
  float* out = (float*)d_out;

  char* ws = (char*)d_ws;
  float*  nrm = (float*)ws;                                   //  64 KB
  float*  gem = (float*)(ws + (64 << 10));                    // 128 KB
  __bf16* wbf = (__bf16*)(ws + (64 << 10) + (128 << 10));     //   2 MB
  __bf16* At  = (__bf16*)(ws + (64 << 10) + (128 << 10) + (2 << 20)); // 64 MB

  size_t projLds = 2 * STAGE_ELEMS * sizeof(__bf16);          // ~92 KB dynamic

  k_prep <<<4096, 256, 0, stream>>>(w, wbf, gem, nrm);
  k_norm1<<<512,  256, 0, stream>>>(x, nrm);
  k_norm2<<<64,   256, 0, stream>>>(nrm);
  k_box  <<<4096, 256, 0, stream>>>(x, nrm, p, At, gem);
  k_gem  <<<16,   256, 0, stream>>>(gem, w, bb, p, out);
  k_proj <<<256,  256, projLds, stream>>>(At, wbf, bb, out + 8192);
}
